// GRU_87703232184583
// MI455X (gfx1250) — compile-verified, matrix operand order checked by reading, not run on hardware
//
#include <hip/hip_runtime.h>
#include <hip/hip_bf16.h>
#include <math.h>

typedef __bf16 bf16_t;
typedef __attribute__((ext_vector_type(16))) __bf16 v16bf;
typedef __attribute__((ext_vector_type(8)))  float  v8f;

#define B_    64
#define S_    1024
#define E_    256
#define H_    512
#define NCOMB 2048            // 2H (gates) + H (gi) + H (gh)
#define K0    768             // E + H   (layer 0 combined input)
#define K1    1024            // H + H   (layer 1 combined input)
#define NUM_WG 16
#define TPB    256            // 8 wave32 per workgroup
#define NWAVES (NUM_WG * (TPB / 32))   // 128 == NCOMB/16 column strips

__device__ __forceinline__ float sigf(float x) { return 1.0f / (1.0f + __expf(-x)); }

// ---------------- device-wide barrier for the persistent kernel ----------------
__device__ __forceinline__ void grid_sync(unsigned* cnt, unsigned* gen) {
  __syncthreads();
  if (threadIdx.x == 0) {
    __threadfence();
    unsigned g = __atomic_load_n(gen, __ATOMIC_RELAXED);
    unsigned a = __atomic_fetch_add(cnt, 1u, __ATOMIC_ACQ_REL);
    if (a == NUM_WG - 1) {
      __atomic_store_n(cnt, 0u, __ATOMIC_RELAXED);
      __atomic_fetch_add(gen, 1u, __ATOMIC_RELEASE);
    } else {
      while (__atomic_load_n(gen, __ATOMIC_ACQUIRE) == g)
        __builtin_amdgcn_s_sleep(1);
    }
    __threadfence();
  }
  __syncthreads();
}

// ---------------- fused-layer GEMM: G[64 x 2048] = X[64 x K] * W^T --------------
// Stage X into LDS once per block (all 8 waves share the A fragments), then each
// wave owns exactly one 16-wide column strip (nt == global wave id, 128 strips ==
// 128 waves) and runs all four 16-row M tiles, reusing each B fragment 4x.
// W is row-major [2048][K] so a lane's B fragment is one contiguous 32B load.
__device__ __forceinline__ void gemm_tiles(const bf16_t* __restrict__ Xg, int K,
                                           const bf16_t* __restrict__ W,
                                           float* __restrict__ G,
                                           bf16_t* __restrict__ Xl /* LDS */) {
  // cooperative copy: global X -> LDS (b128 loads -> ds_store_b128)
  {
    const int nchunks = (B_ * K) >> 4;           // 16 bf16 per 32B chunk
    const v16bf* src = (const v16bf*)Xg;
    v16bf* dst = (v16bf*)Xl;
    for (int i = threadIdx.x; i < nchunks; i += TPB) dst[i] = src[i];
  }
  __syncthreads();

  const int lane  = threadIdx.x & 31;
  const int nt    = blockIdx.x * (TPB >> 5) + (threadIdx.x >> 5);  // column strip
  const int nsub  = lane & 15;          // row/col within tile
  const int khalf = (lane >> 4) << 4;   // lanes 16..31 hold the upper K half
  const int nk    = K >> 5;             // K-steps of 32

  const bf16_t* wrow = W + (size_t)(nt * 16 + nsub) * K;
  v8f acc0 = {}, acc1 = {}, acc2 = {}, acc3 = {};
#pragma clang loop unroll_count(4)
  for (int ks = 0; ks < nk; ++ks) {
    const int kb = (ks << 5) + khalf;
    v16bf bfrag = *(const v16bf*)(wrow + kb);
    if (ks + 1 < nk) __builtin_prefetch((const void*)(wrow + kb + 32), 0, 1);
    const bf16_t* xa = Xl + (size_t)nsub * K + kb;
    v16bf a0 = *(const v16bf*)(xa);
    v16bf a1 = *(const v16bf*)(xa + (size_t)16 * K);
    v16bf a2 = *(const v16bf*)(xa + (size_t)32 * K);
    v16bf a3 = *(const v16bf*)(xa + (size_t)48 * K);
    acc0 = __builtin_amdgcn_wmma_f32_16x16x32_bf16(false, a0, false, bfrag, (short)0, acc0, false, false);
    acc1 = __builtin_amdgcn_wmma_f32_16x16x32_bf16(false, a1, false, bfrag, (short)0, acc1, false, false);
    acc2 = __builtin_amdgcn_wmma_f32_16x16x32_bf16(false, a2, false, bfrag, (short)0, acc2, false, false);
    acc3 = __builtin_amdgcn_wmma_f32_16x16x32_bf16(false, a3, false, bfrag, (short)0, acc3, false, false);
  }
  // C/D layout: lanes 0-15 hold N=lane, M=vgpr; lanes 16-31 hold M=vgpr+8
  const int ncol = nt * 16 + nsub;
  const int mb   = (lane >> 4) << 3;
  float* g = G + ncol;
#pragma unroll
  for (int v = 0; v < 8; ++v) {
    g[(size_t)(     mb + v) * NCOMB] = acc0[v];
    g[(size_t)(16 + mb + v) * NCOMB] = acc1[v];
    g[(size_t)(32 + mb + v) * NCOMB] = acc2[v];
    g[(size_t)(48 + mb + v) * NCOMB] = acc3[v];
  }
}

// ---------------- persistent GRU kernel ----------------
__global__ __launch_bounds__(TPB) void gru_persist(
    const int* __restrict__ tokens, const float* __restrict__ emb,
    const bf16_t* __restrict__ W0, const bf16_t* __restrict__ W1,
    const float* __restrict__ bias0, const float* __restrict__ bias1,
    bf16_t* __restrict__ X0, bf16_t* __restrict__ X1,
    float* __restrict__ G0, float* __restrict__ G1,
    float* __restrict__ h0, float* __restrict__ h1,
    unsigned* __restrict__ sync) {
  extern __shared__ bf16_t smem_x[];   // K1*B_ bf16 = 128 KB (CDNA5: 320 KB/WGP)
  const int tid     = blockIdx.x * blockDim.x + threadIdx.x;
  const int gstride = gridDim.x * blockDim.x;
  unsigned* cnt = sync;
  unsigned* gen = sync + 64;   // separate cache line

  // initial embedding gather for t=0 into X0[:, 0:E]
  for (int i = tid; i < B_ * E_; i += gstride) {
    int b = i >> 8, e = i & (E_ - 1);
    int tok = tokens[b * S_ + 0];
    X0[(size_t)b * K0 + e] = (bf16_t)emb[(size_t)tok * E_ + e];
  }
  grid_sync(cnt, gen);

  for (int t = 0; t < S_; ++t) {
    gemm_tiles(X0, K0, W0, G0, smem_x);           // layer 0 fused GEMM
    grid_sync(cnt, gen);

    // pointwise layer 0: r,z,n -> h0'; feed h0' into both layers' inputs (bf16)
    for (int i = tid; i < B_ * H_; i += gstride) {
      int b = i >> 9, j = i & (H_ - 1);
      const float* g = G0 + (size_t)b * NCOMB;
      float r  = sigf(g[j] + bias0[j]);
      float z  = sigf(g[H_ + j] + bias0[H_ + j]);
      float n  = tanhf(g[2 * H_ + j] + bias0[2 * H_ + j] +
                       r * (g[3 * H_ + j] + bias0[3 * H_ + j]));
      float hn = (1.0f - z) * n + z * h0[(size_t)b * H_ + j];
      h0[(size_t)b * H_ + j]        = hn;
      X0[(size_t)b * K0 + E_ + j]   = (bf16_t)hn;   // next step, layer 0 hidden
      X1[(size_t)b * K1 + j]        = (bf16_t)hn;   // this step, layer 1 input
    }
    grid_sync(cnt, gen);

    gemm_tiles(X1, K1, W1, G1, smem_x);           // layer 1 fused GEMM
    grid_sync(cnt, gen);

    // pointwise layer 1, fused with next step's embedding gather
    for (int i = tid; i < B_ * H_; i += gstride) {
      int b = i >> 9, j = i & (H_ - 1);
      const float* g = G1 + (size_t)b * NCOMB;
      float r  = sigf(g[j] + bias1[j]);
      float z  = sigf(g[H_ + j] + bias1[H_ + j]);
      float n  = tanhf(g[2 * H_ + j] + bias1[2 * H_ + j] +
                       r * (g[3 * H_ + j] + bias1[3 * H_ + j]));
      float hn = (1.0f - z) * n + z * h1[(size_t)b * H_ + j];
      h1[(size_t)b * H_ + j]      = hn;
      X1[(size_t)b * K1 + H_ + j] = (bf16_t)hn;     // next step, layer 1 hidden
    }
    if (t + 1 < S_) {
      for (int i = tid; i < B_ * E_; i += gstride) {
        int b = i >> 8, e = i & (E_ - 1);
        int tok = tokens[b * S_ + (t + 1)];
        X0[(size_t)b * K0 + e] = (bf16_t)emb[(size_t)tok * E_ + e];
      }
    }
    grid_sync(cnt, gen);
  }
}

// ---------------- weight prep: fp32 -> bf16, fuse [gate | gi | gh] ----------------
__global__ void prep_kernel(
    const float* __restrict__ gw0, const float* __restrict__ gi0, const float* __restrict__ gh0,
    const float* __restrict__ gw1, const float* __restrict__ gi1, const float* __restrict__ gh1,
    const float* __restrict__ gb0, const float* __restrict__ gib0, const float* __restrict__ ghb0,
    const float* __restrict__ gb1, const float* __restrict__ gib1, const float* __restrict__ ghb1,
    bf16_t* __restrict__ W0, bf16_t* __restrict__ W1,
    float* __restrict__ bias0, float* __restrict__ bias1,
    bf16_t* __restrict__ X0, bf16_t* __restrict__ X1,
    float* __restrict__ h0, float* __restrict__ h1, unsigned* __restrict__ sync) {
  const int tid = blockIdx.x * blockDim.x + threadIdx.x;
  const int gs  = gridDim.x * blockDim.x;
  for (int i = tid; i < NCOMB * K0; i += gs) {          // layer 0, K = E+H
    int r = i / K0, k = i - r * K0;
    float v;
    if      (r < 1024) v = gw0[(size_t)r * K0 + k];
    else if (r < 1536) v = (k <  E_) ? gi0[(size_t)(r - 1024) * E_ + k] : 0.0f;
    else               v = (k >= E_) ? gh0[(size_t)(r - 1536) * H_ + (k - E_)] : 0.0f;
    W0[i] = (bf16_t)v;
  }
  for (int i = tid; i < NCOMB * K1; i += gs) {          // layer 1, K = H+H
    int r = i / K1, k = i - r * K1;
    float v;
    if      (r < 1024) v = gw1[(size_t)r * K1 + k];
    else if (r < 1536) v = (k <  H_) ? gi1[(size_t)(r - 1024) * H_ + k] : 0.0f;
    else               v = (k >= H_) ? gh1[(size_t)(r - 1536) * H_ + (k - H_)] : 0.0f;
    W1[i] = (bf16_t)v;
  }
  for (int i = tid; i < NCOMB; i += gs) {
    bias0[i] = (i < 1024) ? gb0[i] : (i < 1536 ? gib0[i - 1024] : ghb0[i - 1536]);
    bias1[i] = (i < 1024) ? gb1[i] : (i < 1536 ? gib1[i - 1024] : ghb1[i - 1536]);
  }
  for (int i = tid; i < B_ * K0; i += gs) X0[i] = (bf16_t)0.0f;
  for (int i = tid; i < B_ * K1; i += gs) X1[i] = (bf16_t)0.0f;
  for (int i = tid; i < B_ * H_; i += gs) { h0[i] = 0.0f; h1[i] = 0.0f; }
  for (int i = tid; i < 128; i += gs) sync[i] = 0u;
}

// ---------------- final FC + sigmoid ----------------
__global__ void fc_kernel(const float* __restrict__ h1, const float* __restrict__ fc_w,
                          const float* __restrict__ fc_b, float* __restrict__ out) {
  int b = threadIdx.x;
  if (b < B_) {
    float s = fc_b[0];
    for (int k = 0; k < H_; ++k) s += h1[(size_t)b * H_ + k] * fc_w[k];
    out[b] = 1.0f / (1.0f + __expf(-s));
  }
}

extern "C" void kernel_launch(void* const* d_in, const int* in_sizes, int n_in,
                              void* d_out, int out_size, void* d_ws, size_t ws_size,
                              hipStream_t stream) {
  const int*   tokens = (const int*)  d_in[0];
  const float* emb    = (const float*)d_in[1];
  const float* fc_w   = (const float*)d_in[2];
  const float* fc_b   = (const float*)d_in[3];
  const float* gw0  = (const float*)d_in[4];
  const float* gb0  = (const float*)d_in[5];
  const float* gi0  = (const float*)d_in[6];
  const float* gib0 = (const float*)d_in[7];
  const float* gh0  = (const float*)d_in[8];
  const float* ghb0 = (const float*)d_in[9];
  const float* gw1  = (const float*)d_in[10];
  const float* gb1  = (const float*)d_in[11];
  const float* gi1  = (const float*)d_in[12];
  const float* gib1 = (const float*)d_in[13];
  const float* gh1  = (const float*)d_in[14];
  const float* ghb1 = (const float*)d_in[15];

  char* ws = (char*)d_ws;
  size_t off = 0;
  auto carve = [&](size_t bytes) -> void* {
    void* p = ws + off;
    off += (bytes + 255) & ~(size_t)255;
    return p;
  };
  bf16_t* W0    = (bf16_t*)carve((size_t)NCOMB * K0 * sizeof(bf16_t));  // 3.0 MB
  bf16_t* W1    = (bf16_t*)carve((size_t)NCOMB * K1 * sizeof(bf16_t));  // 4.0 MB
  float*  bias0 = (float*) carve((size_t)NCOMB * sizeof(float));
  float*  bias1 = (float*) carve((size_t)NCOMB * sizeof(float));
  bf16_t* X0    = (bf16_t*)carve((size_t)B_ * K0 * sizeof(bf16_t));
  bf16_t* X1    = (bf16_t*)carve((size_t)B_ * K1 * sizeof(bf16_t));
  float*  G0    = (float*) carve((size_t)B_ * NCOMB * sizeof(float));
  float*  G1    = (float*) carve((size_t)B_ * NCOMB * sizeof(float));
  float*  h0    = (float*) carve((size_t)B_ * H_ * sizeof(float));
  float*  h1    = (float*) carve((size_t)B_ * H_ * sizeof(float));
  unsigned* syn = (unsigned*)carve(128 * sizeof(unsigned));

  prep_kernel<<<512, 256, 0, stream>>>(gw0, gi0, gh0, gw1, gi1, gh1,
                                       gb0, gib0, ghb0, gb1, gib1, ghb1,
                                       W0, W1, bias0, bias1, X0, X1, h0, h1, syn);
  const size_t shbytes = (size_t)B_ * K1 * sizeof(bf16_t);   // 128 KB dynamic LDS
  gru_persist<<<NUM_WG, TPB, shbytes, stream>>>(tokens, emb, W0, W1, bias0, bias1,
                                                X0, X1, G0, G1, h0, h1, syn);
  fc_kernel<<<1, 64, 0, stream>>>(h1, fc_w, fc_b, (float*)d_out);
}
